// FP8Attention_17420387352696
// MI455X (gfx1250) — compile-verified, matrix operand order checked by reading, not run on hardware
//
#include <hip/hip_runtime.h>

// ---------------------------------------------------------------------------
// Types for CDNA5 (gfx1250) wave32 WMMA
// ---------------------------------------------------------------------------
typedef __bf16 v16bf __attribute__((ext_vector_type(16)));
typedef __bf16 bf8v  __attribute__((ext_vector_type(8)));
typedef float  v8f   __attribute__((ext_vector_type(8)));
typedef int    v4i   __attribute__((vector_size(16)));  // matches async builtin param

static constexpr int HIDDEN   = 2048;
static constexpr int NHEADS   = 16;
static constexpr int NKVHEADS = 4;
static constexpr int HEADDIM  = 128;
static constexpr int KVDIM    = NKVHEADS * HEADDIM;  // 512
static constexpr int Bsz      = 2;
static constexpr int Ssz      = 2048;
static constexpr int BS       = Bsz * Ssz;           // 4096

__device__ __forceinline__ __bf16 f2bf(float f) {
  union { float f; unsigned u; } a; a.f = f;
  unsigned r = (a.u + 0x7FFFu + ((a.u >> 16) & 1u)) >> 16;
  union { unsigned short s; __bf16 b; } o; o.s = (unsigned short)r;
  return o.b;
}
__device__ __forceinline__ float bf2f(__bf16 b) {
  union { unsigned short s; __bf16 b; } i; i.b = b;
  union { unsigned u; float f; } o; o.u = ((unsigned)i.s) << 16;
  return o.f;
}

__device__ __forceinline__ v8f wmma_bf16(v16bf a, v16bf b, v8f c) {
  // D = A(16x32 bf16) * B(32x16 bf16) + C(16x16 f32)
  return __builtin_amdgcn_wmma_f32_16x16x32_bf16(
      false, a, false, b, (short)0, c, false, false);
}

// Build a 16x32 A-fragment (or 32x16 B-fragment, symmetric role) from a
// row-major tile (global or LDS).  rows = M or N (lane&15), cols = K.
// Wave32 16-bit layout: lane<16 holds K = {0..7, 16..23}, lane>=16 holds
// K = {8..15, 24..31}; two contiguous 16-byte loads per lane.
__device__ __forceinline__ v16bf load_frag(const __bf16* tile, int ld, int lane) {
  int row  = lane & 15;
  int koff = ((lane >> 4) & 1) * 8;
  const __bf16* p = tile + (size_t)row * ld + koff;
  bf8v lo = *(const bf8v*)(p);
  bf8v hi = *(const bf8v*)(p + 16);
  v16bf r;
#pragma unroll
  for (int i = 0; i < 8; ++i) { r[i] = lo[i]; r[i + 8] = hi[i]; }
  return r;
}

// ---------------------------------------------------------------------------
// CDNA5 async global->LDS copy (ASYNCcnt-tracked DMA path), with sync fallback
// ---------------------------------------------------------------------------
#if defined(__HIP_DEVICE_COMPILE__) && \
    __has_builtin(__builtin_amdgcn_global_load_async_to_lds_b128) && \
    __has_builtin(__builtin_amdgcn_s_wait_asynccnt)
#define CDNA5_ASYNC_LDS 1
#else
#define CDNA5_ASYNC_LDS 0
#endif

#if CDNA5_ASYNC_LDS
typedef __attribute__((address_space(1))) v4i* gptr_v4i;
typedef __attribute__((address_space(3))) v4i* lptr_v4i;
#endif

__device__ __forceinline__ void copy16_g2l(__bf16* lds, const __bf16* g) {
#if CDNA5_ASYNC_LDS
  __builtin_amdgcn_global_load_async_to_lds_b128(
      (gptr_v4i)(g), (lptr_v4i)(lds), 0, 0);
#else
  *(bf8v*)lds = *(const bf8v*)g;
#endif
}

template <int N>
__device__ __forceinline__ void wait_async() {
#if CDNA5_ASYNC_LDS
  __builtin_amdgcn_s_wait_asynccnt(N);
#endif
}

// ---------------------------------------------------------------------------
// fp32 -> bf16 conversion
// ---------------------------------------------------------------------------
__global__ void cvt_f32_bf16(const float* __restrict__ src,
                             __bf16* __restrict__ dst, int n) {
  int i = blockIdx.x * blockDim.x + threadIdx.x;
  if (i < n) dst[i] = f2bf(src[i]);
}

// ---------------------------------------------------------------------------
// GEMM: C[M,N] = A[M,K] (row major) * W[N,K]^T (W row major, K contiguous)
// Block = 256 threads = 8 waves (4 along M x 2 along N).
// Block tile 128x64, wave tile 32x32 (4 accumulators).
// K-slabs (A:128x32, W:64x32) double-buffered in LDS, filled with
// global_load_async_to_lds_b128 and drained with s_wait_asynccnt.
// ---------------------------------------------------------------------------
template <bool OUT_F32>
__global__ __launch_bounds__(256) void gemm_bf16_nt(
    const __bf16* __restrict__ A, const __bf16* __restrict__ W,
    void* __restrict__ C, int M, int N, int K) {
  __shared__ __align__(16) __bf16 Asb[2][128][32];  // 2 x 8 KB
  __shared__ __align__(16) __bf16 Bsb[2][64][32];   // 2 x 4 KB

  const int tid  = threadIdx.x;
  const int lane = tid & 31;
  const int wave = tid >> 5;
  const int half = (lane >> 4) & 1;
  const int l16  = lane & 15;
  const int wm   = (wave & 3) * 32;   // wave M offset in block tile
  const int wn   = (wave >> 2) * 32;  // wave N offset in block tile
  const int mblk = blockIdx.y * 128;
  const int nblk = blockIdx.x * 64;

  // Staging map: A = 512 16B chunks (2/thread), B = 256 16B chunks (1/thread)
  // 128 rows x 4 chunks/row = 512 chunks: chunk c -> row c>>2, col (c&3)*8
  const int ca0 = tid, ca1 = tid + 256;
  const int arow0 = ca0 >> 2, acol0 = (ca0 & 3) * 8;
  const int arow1 = ca1 >> 2, acol1 = (ca1 & 3) * 8;
  const int brow  = tid >> 2, bcol  = (tid & 3) * 8;

  auto stage = [&](int k0, int buf) {
    copy16_g2l(&Asb[buf][arow0][acol0], A + (size_t)(mblk + arow0) * K + k0 + acol0);
    copy16_g2l(&Asb[buf][arow1][acol1], A + (size_t)(mblk + arow1) * K + k0 + acol1);
    copy16_g2l(&Bsb[buf][brow][bcol],   W + (size_t)(nblk + brow) * K + k0 + bcol);
  };

  v8f acc00 = 0.0f, acc01 = 0.0f, acc10 = 0.0f, acc11 = 0.0f;

  const int nk = K / 32;
  stage(0, 0);
  for (int ki = 0; ki < nk; ++ki) {
    const int buf = ki & 1;
    if (ki + 1 < nk) {
      stage((ki + 1) * 32, buf ^ 1);  // prefetch next slab (async)
      wait_async<3>();                // our 3 loads for slab ki are done
    } else {
      wait_async<0>();
    }
    __syncthreads();                  // everyone's slab ki is resident

    v16bf a0 = load_frag(&Asb[buf][wm][0],      32, lane);
    v16bf a1 = load_frag(&Asb[buf][wm + 16][0], 32, lane);
    v16bf b0 = load_frag(&Bsb[buf][wn][0],      32, lane);
    v16bf b1 = load_frag(&Bsb[buf][wn + 16][0], 32, lane);
    acc00 = wmma_bf16(a0, b0, acc00);
    acc01 = wmma_bf16(a0, b1, acc01);
    acc10 = wmma_bf16(a1, b0, acc10);
    acc11 = wmma_bf16(a1, b1, acc11);

    __syncthreads();                  // done reading slab ki; safe to overwrite
  }

#pragma unroll
  for (int r = 0; r < 8; ++r) {
    int row0 = mblk + wm + r + 8 * half;        // C/D layout: M = r + 8*(lane>=16)
    int row1 = row0 + 16;
    int col0 = nblk + wn + l16;
    int col1 = col0 + 16;
    if (OUT_F32) {
      float* Cf = (float*)C;
      Cf[(size_t)row0 * N + col0] = acc00[r];
      Cf[(size_t)row0 * N + col1] = acc01[r];
      Cf[(size_t)row1 * N + col0] = acc10[r];
      Cf[(size_t)row1 * N + col1] = acc11[r];
    } else {
      __bf16* Cb = (__bf16*)C;
      Cb[(size_t)row0 * N + col0] = f2bf(acc00[r]);
      Cb[(size_t)row0 * N + col1] = f2bf(acc01[r]);
      Cb[(size_t)row1 * N + col0] = f2bf(acc10[r]);
      Cb[(size_t)row1 * N + col1] = f2bf(acc11[r]);
    }
  }
}

// ---------------------------------------------------------------------------
// RoPE in place on bf16 [BS, nheads*128]; thread = (bs, head, d<64)
// ---------------------------------------------------------------------------
__global__ void rope_kernel(__bf16* __restrict__ x, int nheads, int total) {
  int idx = blockIdx.x * blockDim.x + threadIdx.x;
  if (idx >= total) return;
  int d  = idx & 63;
  int h  = (idx >> 6) % nheads;
  int bs = idx / (64 * nheads);
  int pos = bs & (Ssz - 1);
  __bf16* p = x + (size_t)bs * nheads * HEADDIM + h * HEADDIM;
  float x0 = bf2f(p[d]);
  float x1 = bf2f(p[d + 64]);
  float inv = __powf(10000.0f, -2.0f * (float)d / 128.0f);
  float ang = (float)pos * inv;
  float s, c;
  __sincosf(ang, &s, &c);
  p[d]      = f2bf(x0 * c - x1 * s);
  p[d + 64] = f2bf(x1 * c + x0 * s);
}

// ---------------------------------------------------------------------------
// Flash attention (causal + pad mask), GQA 16 Q heads / 4 KV heads.
// Grid: (S/64, NHEADS, B).  Block: 128 threads = 4 waves, wave = 16 q-rows.
// Key chunks of 32; scores via WMMA over D=128; P -> LDS -> A-frag; PV via
// WMMA with V^T staged in LDS.
// ---------------------------------------------------------------------------
__global__ __launch_bounds__(128) void flash_attn(
    const __bf16* __restrict__ Q, const __bf16* __restrict__ Kc,
    const __bf16* __restrict__ Vc, const int* __restrict__ amask,
    __bf16* __restrict__ Ctx) {
  __shared__ __align__(16) __bf16 VT[HEADDIM][32];     // V^T chunk, 8 KB
  __shared__ __align__(16) __bf16 Plds[4][16][32];     // per-wave P, 4 KB

  const int wave = threadIdx.x >> 5;
  const int lane = threadIdx.x & 31;
  const int half = (lane >> 4) & 1;
  const int l16  = lane & 15;
  const int b  = blockIdx.z;
  const int h  = blockIdx.y;
  const int q0 = blockIdx.x * 64;
  const int kvh = h / (NHEADS / NKVHEADS);
  const int qrow0 = q0 + wave * 16;
  const float scale = 0.08838834764831845f;  // 1/sqrt(128)

  const __bf16* Qrow  = Q  + ((size_t)(b * Ssz + qrow0)) * HIDDEN + h * HEADDIM;
  const __bf16* Kbase = Kc + ((size_t)(b * Ssz)) * KVDIM + kvh * HEADDIM;
  const __bf16* Vbase = Vc + ((size_t)(b * Ssz)) * KVDIM + kvh * HEADDIM;

  v16bf qf[4];
#pragma unroll
  for (int t = 0; t < 4; ++t) qf[t] = load_frag(Qrow + t * 32, HIDDEN, lane);

  v8f o[8];
#pragma unroll
  for (int t = 0; t < 8; ++t) o[t] = 0.0f;
  float m_i[8], l_i[8];
#pragma unroll
  for (int r = 0; r < 8; ++r) { m_i[r] = -3.0e30f; l_i[r] = 0.0f; }

  const int kend = q0 + 64;   // causal upper bound for this q tile
  for (int kc = 0; kc < kend; kc += 32) {
    __syncthreads();
    // Stage V^T[d][key] into LDS (coalesced reads, transposed writes)
    for (int i = threadIdx.x; i < 32 * 16; i += 128) {
      int kr = i >> 4;
      int cb = (i & 15) * 8;
      bf8v v = *(const bf8v*)(Vbase + (size_t)(kc + kr) * KVDIM + cb);
#pragma unroll
      for (int j = 0; j < 8; ++j) VT[cb + j][kr] = v[j];
    }
    __syncthreads();

    // ---- scores S = Q * K^T for 16 q-rows x 32 keys ----
    v8f s0 = 0.0f, s1 = 0.0f;
#pragma unroll
    for (int t = 0; t < 4; ++t) {
      v16bf kb0 = load_frag(Kbase + (size_t)kc * KVDIM + t * 32, KVDIM, lane);
      v16bf kb1 = load_frag(Kbase + (size_t)(kc + 16) * KVDIM + t * 32, KVDIM, lane);
      s0 = wmma_bf16(qf[t], kb0, s0);
      s1 = wmma_bf16(qf[t], kb1, s1);
    }

    // pad mask for this lane's two key columns (same for all rows)
    const int col0 = kc + l16;
    const int col1 = kc + 16 + l16;
    const bool pad0 = (amask[b * Ssz + col0] == 0);
    const bool pad1 = (amask[b * Ssz + col1] == 0);

    // ---- online softmax + P to LDS ----
#pragma unroll
    for (int r = 0; r < 8; ++r) {
      const int qidx = qrow0 + r + 8 * half;
      float e0 = s0[r] * scale;
      float e1 = s1[r] * scale;
      if (col0 > qidx || pad0) e0 = -3.0e30f;
      if (col1 > qidx || pad1) e1 = -3.0e30f;
      float mx = fmaxf(e0, e1);
#pragma unroll
      for (int off = 1; off < 16; off <<= 1)
        mx = fmaxf(mx, __shfl_xor(mx, off, 32));
      float m_new = fmaxf(m_i[r], mx);
      float p0 = __expf(e0 - m_new);
      float p1 = __expf(e1 - m_new);
      float rs = p0 + p1;
#pragma unroll
      for (int off = 1; off < 16; off <<= 1)
        rs += __shfl_xor(rs, off, 32);
      float alpha = __expf(m_i[r] - m_new);
      l_i[r] = l_i[r] * alpha + rs;
      m_i[r] = m_new;
#pragma unroll
      for (int t = 0; t < 8; ++t) o[t][r] *= alpha;
      Plds[wave][r + 8 * half][l16]      = f2bf(p0);
      Plds[wave][r + 8 * half][16 + l16] = f2bf(p1);
    }

    // ---- reload P as an A-fragment (wave-local LDS, in-order with DS) ----
    {
      const __bf16* pp = &Plds[wave][l16][half * 8];
      bf8v lo = *(const bf8v*)(pp);
      bf8v hi = *(const bf8v*)(pp + 16);
      v16bf pa;
#pragma unroll
      for (int i = 0; i < 8; ++i) { pa[i] = lo[i]; pa[i + 8] = hi[i]; }

      // ---- O += P(16x32) * V(32x128) over 8 d-tiles ----
#pragma unroll
      for (int t = 0; t < 8; ++t) {
        const __bf16* vp = &VT[t * 16 + l16][half * 8];
        bf8v vlo = *(const bf8v*)(vp);
        bf8v vhi = *(const bf8v*)(vp + 16);
        v16bf vb;
#pragma unroll
        for (int i = 0; i < 8; ++i) { vb[i] = vlo[i]; vb[i + 8] = vhi[i]; }
        o[t] = wmma_bf16(pa, vb, o[t]);
      }
    }
  }

  // ---- normalize and store context [BS, HIDDEN] ----
#pragma unroll
  for (int r = 0; r < 8; ++r) {
    float invl = 1.0f / l_i[r];
    int qidx = qrow0 + r + 8 * half;
    __bf16* dst = Ctx + ((size_t)(b * Ssz + qidx)) * HIDDEN + h * HEADDIM;
#pragma unroll
    for (int t = 0; t < 8; ++t) dst[t * 16 + l16] = f2bf(o[t][r] * invl);
  }
}

// ---------------------------------------------------------------------------
// Host-side launch
// ---------------------------------------------------------------------------
extern "C" void kernel_launch(void* const* d_in, const int* in_sizes, int n_in,
                              void* d_out, int out_size, void* d_ws, size_t ws_size,
                              hipStream_t stream) {
  const float* hx    = (const float*)d_in[0];
  const int*   amask = (const int*)d_in[1];
  const float* wq    = (const float*)d_in[2];
  const float* wk    = (const float*)d_in[3];
  const float* wv    = (const float*)d_in[4];
  const float* wo    = (const float*)d_in[5];

  char* ws = (char*)d_ws;
  size_t off = 0;
  auto alloc = [&](size_t bytes) -> __bf16* {
    __bf16* p = (__bf16*)(ws + off);
    off += (bytes + 255) & ~size_t(255);
    return p;
  };
  __bf16* Xb  = alloc((size_t)BS * HIDDEN * 2);
  __bf16* Wqb = alloc((size_t)HIDDEN * HIDDEN * 2);
  __bf16* Wkb = alloc((size_t)KVDIM * HIDDEN * 2);
  __bf16* Wvb = alloc((size_t)KVDIM * HIDDEN * 2);
  __bf16* Wob = alloc((size_t)HIDDEN * HIDDEN * 2);
  __bf16* Qb  = alloc((size_t)BS * HIDDEN * 2);
  __bf16* Kb  = alloc((size_t)BS * KVDIM * 2);
  __bf16* Vb  = alloc((size_t)BS * KVDIM * 2);
  __bf16* Ctx = alloc((size_t)BS * HIDDEN * 2);

  auto cvt = [&](const float* s, __bf16* d, int n) {
    cvt_f32_bf16<<<(n + 255) / 256, 256, 0, stream>>>(s, d, n);
  };
  cvt(hx, Xb,  BS * HIDDEN);
  cvt(wq, Wqb, HIDDEN * HIDDEN);
  cvt(wk, Wkb, KVDIM * HIDDEN);
  cvt(wv, Wvb, KVDIM * HIDDEN);
  cvt(wo, Wob, HIDDEN * HIDDEN);

  // Q/K/V projections (bf16 out)
  gemm_bf16_nt<false><<<dim3(HIDDEN / 64, BS / 128), 256, 0, stream>>>(
      Xb, Wqb, Qb, BS, HIDDEN, HIDDEN);
  gemm_bf16_nt<false><<<dim3(KVDIM / 64, BS / 128), 256, 0, stream>>>(
      Xb, Wkb, Kb, BS, KVDIM, HIDDEN);
  gemm_bf16_nt<false><<<dim3(KVDIM / 64, BS / 128), 256, 0, stream>>>(
      Xb, Wvb, Vb, BS, KVDIM, HIDDEN);

  // RoPE in place on Q and K
  {
    int tq = BS * NHEADS * 64;
    rope_kernel<<<(tq + 255) / 256, 256, 0, stream>>>(Qb, NHEADS, tq);
    int tk = BS * NKVHEADS * 64;
    rope_kernel<<<(tk + 255) / 256, 256, 0, stream>>>(Kb, NKVHEADS, tk);
  }

  // Flash attention -> context (bf16)
  flash_attn<<<dim3(Ssz / 64, NHEADS, Bsz), 128, 0, stream>>>(
      Qb, Kb, Vb, amask, Ctx);

  // Output projection (f32 out)
  gemm_bf16_nt<true><<<dim3(HIDDEN / 64, BS / 128), 256, 0, stream>>>(
      Ctx, Wob, d_out, BS, HIDDEN, HIDDEN);
}
